// TransformerBlock_16071767622398
// MI455X (gfx1250) — matrix operand order repaired and checked by reading, not compiled, not run on hardware
//
#include <hip/hip_runtime.h>
#include <hip/hip_bf16.h>
#include <math.h>

// ---------------------------------------------------------------------------
// Feature probes (compile-safe: fall back to sync copies if builtins absent)
// ---------------------------------------------------------------------------
#ifndef __has_builtin
#define __has_builtin(x) 0
#endif

#if __has_builtin(__builtin_amdgcn_global_load_async_to_lds_b128)
#define HAVE_ASYNC_LDS 1
#else
#define HAVE_ASYNC_LDS 0
#endif

#if HAVE_ASYNC_LDS && __has_builtin(__builtin_amdgcn_s_wait_asynccnt)
#define WAIT_ASYNC() __builtin_amdgcn_s_wait_asynccnt(0)
#elif HAVE_ASYNC_LDS
#define WAIT_ASYNC() asm volatile("s_wait_asynccnt 0x0" ::: "memory")
#else
#define WAIT_ASYNC()
#endif

// Pointer types matching the async builtin's signature:
//   param0: v4i in global (__device__, AS1) address space
//   param1: v4i in LDS (__shared__, AS3) address space
typedef int v4i_ __attribute__((vector_size(16)));
typedef __attribute__((address_space(1))) v4i_ as1_v4i;
typedef __attribute__((address_space(3))) v4i_ as3_v4i;

// 16-byte global -> LDS copy; async (ASYNCcnt-tracked) when available.
__device__ __forceinline__ void cp16_g2l(const unsigned short* g, unsigned short* l) {
#if HAVE_ASYNC_LDS
    __builtin_amdgcn_global_load_async_to_lds_b128((as1_v4i*)g, (as3_v4i*)l, 0, 0);
#else
    *(uint4*)l = *(const uint4*)g;
#endif
}

// ---------------------------------------------------------------------------
// Types for WMMA fragments (CDNA5 gfx1250, wave32)
// ---------------------------------------------------------------------------
typedef __attribute__((ext_vector_type(16))) __bf16 v16bf;
typedef __attribute__((ext_vector_type(8)))  float  v8f;

union Frag {           // 32 bytes = 8 VGPRs = one 16x32 bf16 A or 32x16 bf16 B fragment
    uint4  u[2];
    v16bf  h;
};

__device__ __forceinline__ v8f wmma_bf16(const Frag& a, const Frag& b, v8f c) {
    return __builtin_amdgcn_wmma_f32_16x16x32_bf16(
        /*neg_a=*/false, a.h, /*neg_b=*/false, b.h,
        /*c_mod=*/(short)0, c, /*reuse_a=*/false, /*reuse_b=*/false);
}

__device__ __forceinline__ unsigned short f2b(float f) {   // fp32 -> bf16 RNE
    unsigned int u = __float_as_uint(f);
    unsigned int r = (u + 0x7FFFu + ((u >> 16) & 1u)) >> 16;
    return (unsigned short)r;
}

// ---------------------------------------------------------------------------
// Problem constants
// ---------------------------------------------------------------------------
#define B_      4
#define N_      2048
#define D_      1024
#define D3_     3072
#define H_      16
#define HD_     64
#define MLP_    4096
#define TOK_    (B_ * N_)   // 8192

// ---------------------------------------------------------------------------
// Elementwise cast  x (fp32) -> bf16
// ---------------------------------------------------------------------------
__global__ __launch_bounds__(256) void cast_f32_bf16(const float* __restrict__ x,
                                                     unsigned short* __restrict__ xb,
                                                     int nelem) {
    int i = (blockIdx.x * 256 + threadIdx.x) * 4;
    if (i + 3 < nelem) {
        float4 v = *(const float4*)&x[i];
        unsigned int lo = (unsigned int)f2b(v.x) | ((unsigned int)f2b(v.y) << 16);
        unsigned int hi = (unsigned int)f2b(v.z) | ((unsigned int)f2b(v.w) << 16);
        *(uint2*)&xb[i] = make_uint2(lo, hi);
    }
}

// ---------------------------------------------------------------------------
// Transpose + cast  W[K,N] fp32  ->  Wt[N,K] bf16   (32x32 LDS tiles)
// ---------------------------------------------------------------------------
__global__ __launch_bounds__(256) void transpose_cast(const float* __restrict__ W,
                                                      unsigned short* __restrict__ Wt,
                                                      int K, int N) {
    __shared__ float tile[32][33];
    int n0 = blockIdx.x * 32, k0 = blockIdx.y * 32;
    for (int i = threadIdx.y; i < 32; i += 8)
        tile[i][threadIdx.x] = W[(size_t)(k0 + i) * N + n0 + threadIdx.x];
    __syncthreads();
    for (int i = threadIdx.y; i < 32; i += 8)
        Wt[(size_t)(n0 + i) * K + k0 + threadIdx.x] = f2b(tile[threadIdx.x][i]);
}

// ---------------------------------------------------------------------------
// BF16 WMMA GEMM:  C[M,N] = A[M,K] @ Bt[N,K]^T  (both bf16, fp32 accumulate)
//   MODE 0: store bf16
//   MODE 1: out_f32 = acc + bias[col] + res[row,col]
//   MODE 2: out_bf16 = gelu(acc + bias[col])        gelu(x)=x*sigmoid(1.702x)
//   MODE 3: store bf16, scaling columns < 1024 by 0.125 (QKV: pre-scale Q)
// Block = 256 threads (8 waves, 4x2), tile 128x128, K-step 32,
// double-buffered LDS with async global->LDS loads when available.
// ---------------------------------------------------------------------------
#define LDSPITCH 40   // bf16 elements per LDS tile row (32 data + 8 pad)

template <int MODE>
__global__ __launch_bounds__(256) void gemm_bf16(const unsigned short* __restrict__ A,
                                                 const unsigned short* __restrict__ Bt,
                                                 int M, int N, int K,
                                                 const float* __restrict__ bias,
                                                 const float* __restrict__ res,
                                                 float* __restrict__ outF,
                                                 unsigned short* __restrict__ outB) {
    __shared__ unsigned short As[2][128 * LDSPITCH];
    __shared__ unsigned short Bs[2][128 * LDSPITCH];

    const int tid  = threadIdx.x;
    const int wv   = tid >> 5;
    const int lane = tid & 31;
    const int m16  = lane & 15;
    const int hl   = lane >> 4;          // half-wave select
    const int wM   = wv >> 1;            // 0..3
    const int wN   = wv & 1;             // 0..1
    const int rowBase = blockIdx.y * 128;
    const int colBase = blockIdx.x * 128;

    // stage one 128x32 K-tile of A and Bt into LDS buffer `buf`
    auto stage = [&](int buf, int k0) {
        for (int c = tid; c < 512; c += 256) {
            int r  = c >> 2;
            int cc = (c & 3) << 3;
            cp16_g2l(&A[(size_t)(rowBase + r) * K + k0 + cc],
                     &As[buf][r * LDSPITCH + cc]);
            cp16_g2l(&Bt[(size_t)(colBase + r) * K + k0 + cc],
                     &Bs[buf][r * LDSPITCH + cc]);
        }
    };

    v8f acc[2][4] = {};

    stage(0, 0);
    int buf = 0;
    for (int k0 = 0; k0 < K; k0 += 32, buf ^= 1) {
        WAIT_ASYNC();        // our async writes into As/Bs[buf] have landed
        __syncthreads();     // everyone's writes visible; prev reads of buf^1 done
        if (k0 + 32 < K) stage(buf ^ 1, k0 + 32);   // prefetch next tile

        Frag af[2], bf[4];
        for (int mt = 0; mt < 2; ++mt) {
            int r = (wM * 32 + mt * 16 + m16) * LDSPITCH;
            af[mt].u[0] = *(const uint4*)&As[buf][r + hl * 8];
            af[mt].u[1] = *(const uint4*)&As[buf][r + 16 + hl * 8];
        }
        for (int nt = 0; nt < 4; ++nt) {
            int r = (wN * 64 + nt * 16 + m16) * LDSPITCH;
            bf[nt].u[0] = *(const uint4*)&Bs[buf][r + hl * 8];
            bf[nt].u[1] = *(const uint4*)&Bs[buf][r + 16 + hl * 8];
        }
        for (int mt = 0; mt < 2; ++mt)
            for (int nt = 0; nt < 4; ++nt)
                acc[mt][nt] = wmma_bf16(af[mt], bf[nt], acc[mt][nt]);
    }

    // -- epilogue --
    for (int mt = 0; mt < 2; ++mt) {
        for (int nt = 0; nt < 4; ++nt) {
            int gc = colBase + wN * 64 + nt * 16 + m16;
            float bv = (MODE == 1 || MODE == 2) ? bias[gc] : 0.0f;
            for (int r = 0; r < 8; ++r) {
                int gr = rowBase + wM * 32 + mt * 16 + r + 8 * hl;
                size_t off = (size_t)gr * N + gc;
                float v = acc[mt][nt][r];
                if (MODE == 0) {
                    outB[off] = f2b(v);
                } else if (MODE == 3) {
                    outB[off] = f2b(gc < D_ ? v * 0.125f : v);  // pre-scale Q
                } else if (MODE == 1) {
                    outF[off] = v + bv + res[off];
                } else {
                    v += bv;
                    float g = v / (1.0f + __expf(-1.702f * v));
                    outB[off] = f2b(g);
                }
            }
        }
    }
}

// ---------------------------------------------------------------------------
// Flash-style attention. qkv bf16 [B,N,3072] -> attn bf16 [B,N,1024].
// Q is already scaled by 1/sqrt(hd) (QKV GEMM epilogue, exact power of 2).
// Block = 128 threads (4 waves). Each wave owns 32 query rows; stream keys
// in 64-wide tiles with online softmax.
// ---------------------------------------------------------------------------
#define KPITCH 72   // bf16 elements per LDS row (64 data + 8 pad)
#define SPITCH 66

__global__ __launch_bounds__(128) void attn_kernel(const unsigned short* __restrict__ qkv,
                                                   unsigned short* __restrict__ attn) {
    __shared__ unsigned short Ks[64 * KPITCH];
    __shared__ unsigned short Vt[64 * KPITCH];
    __shared__ float          Sb[4][32][SPITCH];
    __shared__ unsigned short Pb[4][32 * KPITCH];
    __shared__ float          fac[4][32];
    __shared__ float          linv[4][32];

    const int tid  = threadIdx.x;
    const int wv   = tid >> 5;
    const int lane = tid & 31;
    const int m16  = lane & 15;
    const int hl   = lane >> 4;
    const int bh   = blockIdx.y;
    const int bb   = bh >> 4;
    const int head = bh & 15;
    const int q0   = blockIdx.x * 128 + wv * 32;
    const size_t baseTok = (size_t)bb * N_;

    // Q fragments live in registers for the whole kernel
    Frag qf[2][2];
    for (int mt = 0; mt < 2; ++mt)
        for (int ks = 0; ks < 2; ++ks) {
            const unsigned short* p =
                qkv + (baseTok + q0 + mt * 16 + m16) * D3_ + head * HD_ + ks * 32;
            qf[mt][ks].u[0] = *(const uint4*)(p + hl * 8);
            qf[mt][ks].u[1] = *(const uint4*)(p + 16 + hl * 8);
        }

    float m_i = -__builtin_inff();
    float l_i = 0.0f;
    v8f o[2][4] = {};

    for (int j0 = 0; j0 < N_; j0 += 64) {
        // K tile: 64 keys x 64 hd, row-major (async global->LDS when available)
        for (int c = tid; c < 512; c += 128) {
            int r = c >> 3, cc = (c & 7) << 3;
            cp16_g2l(&qkv[(baseTok + j0 + r) * D3_ + D_ + head * HD_ + cc],
                     &Ks[r * KPITCH + cc]);
        }
        // V tile transposed into LDS: Vt[hd][key]
        for (int c = tid; c < 512; c += 128) {
            int r = c >> 3, cc = (c & 7) << 3;
            uint4 u = *(const uint4*)&qkv[(baseTok + j0 + r) * D3_ + 2 * D_ + head * HD_ + cc];
            const unsigned short* e = (const unsigned short*)&u;
            for (int t = 0; t < 8; ++t) Vt[(cc + t) * KPITCH + r] = e[t];
        }
        WAIT_ASYNC();
        __syncthreads();

        // S = Q @ K^T  (2 M-tiles x 4 N-tiles, K = 64 in two steps)
        v8f s[2][4] = {};
        for (int nt = 0; nt < 4; ++nt)
            for (int ks = 0; ks < 2; ++ks) {
                Frag bf;
                int r = (nt * 16 + m16) * KPITCH + ks * 32 + hl * 8;
                bf.u[0] = *(const uint4*)&Ks[r];
                bf.u[1] = *(const uint4*)&Ks[r + 16];
                for (int mt = 0; mt < 2; ++mt)
                    s[mt][nt] = wmma_bf16(qf[mt][ks], bf, s[mt][nt]);
            }

        // Dump S to LDS in row-major fp32 (no scaling: Q pre-scaled)
        for (int mt = 0; mt < 2; ++mt)
            for (int nt = 0; nt < 4; ++nt)
                for (int r = 0; r < 8; ++r)
                    Sb[wv][mt * 16 + r + 8 * hl][nt * 16 + m16] = s[mt][nt][r];
        __syncthreads();

        // Online softmax: lane == row within the wave's 32 rows
        {
            float rmax = -__builtin_inff();
            for (int c = 0; c < 64; ++c) rmax = fmaxf(rmax, Sb[wv][lane][c]);
            float nm = fmaxf(m_i, rmax);
            float f  = __expf(m_i - nm);
            float sum = 0.0f;
            for (int c = 0; c < 64; ++c) {
                float p = __expf(Sb[wv][lane][c] - nm);
                sum += p;
                Pb[wv][lane * KPITCH + c] = f2b(p);
            }
            l_i = l_i * f + sum;
            m_i = nm;
            fac[wv][lane] = f;
        }
        __syncthreads();

        // Rescale running O, then O += P @ V^T
        for (int mt = 0; mt < 2; ++mt) {
            float fr[8];
            for (int r = 0; r < 8; ++r) fr[r] = fac[wv][mt * 16 + r + 8 * hl];
            for (int nt2 = 0; nt2 < 4; ++nt2)
                for (int r = 0; r < 8; ++r) o[mt][nt2][r] *= fr[r];
        }
        for (int nt2 = 0; nt2 < 4; ++nt2)
            for (int ks = 0; ks < 2; ++ks) {
                Frag bf;
                int rb = (nt2 * 16 + m16) * KPITCH + ks * 32 + hl * 8;
                bf.u[0] = *(const uint4*)&Vt[rb];
                bf.u[1] = *(const uint4*)&Vt[rb + 16];
                for (int mt = 0; mt < 2; ++mt) {
                    Frag af;
                    int ra = (mt * 16 + m16) * KPITCH + ks * 32;
                    af.u[0] = *(const uint4*)&Pb[wv][ra + hl * 8];
                    af.u[1] = *(const uint4*)&Pb[wv][ra + 16 + hl * 8];
                    o[mt][nt2] = wmma_bf16(af, bf, o[mt][nt2]);
                }
            }
        __syncthreads();
    }

    linv[wv][lane] = 1.0f / l_i;    // lane == row
    __syncthreads();

    // Store O / l as bf16 into attn [B,N,1024]
    for (int mt = 0; mt < 2; ++mt) {
        float li[8];
        for (int r = 0; r < 8; ++r) li[r] = linv[wv][mt * 16 + r + 8 * hl];
        for (int nt2 = 0; nt2 < 4; ++nt2)
            for (int r = 0; r < 8; ++r) {
                size_t tokrow = baseTok + q0 + mt * 16 + r + 8 * hl;
                attn[tokrow * D_ + head * HD_ + nt2 * 16 + m16] = f2b(o[mt][nt2][r] * li[r]);
            }
    }
}

// ---------------------------------------------------------------------------
// LayerNorm over rows of 1024; writes fp32 and optionally bf16 copy.
// ---------------------------------------------------------------------------
__global__ __launch_bounds__(256) void ln_kernel(const float* __restrict__ y,
                                                 const float* __restrict__ g,
                                                 const float* __restrict__ be,
                                                 float* __restrict__ outF,
                                                 unsigned short* __restrict__ outB) {
    __shared__ float red[256];
    const int row = blockIdx.x;
    const int tid = threadIdx.x;
    const float* p = y + (size_t)row * D_;

    float s = 0.0f, s2 = 0.0f;
    for (int c = tid; c < D_; c += 256) {
        float v = p[c];
        s += v;
        s2 += v * v;
    }
    red[tid] = s;
    __syncthreads();
    for (int w = 128; w > 0; w >>= 1) {
        if (tid < w) red[tid] += red[tid + w];
        __syncthreads();
    }
    float mean = red[0] * (1.0f / D_);
    __syncthreads();
    red[tid] = s2;
    __syncthreads();
    for (int w = 128; w > 0; w >>= 1) {
        if (tid < w) red[tid] += red[tid + w];
        __syncthreads();
    }
    float var  = red[0] * (1.0f / D_) - mean * mean;
    float rstd = rsqrtf(var + 1e-5f);

    for (int c = tid; c < D_; c += 256) {
        float v = (p[c] - mean) * rstd * g[c] + be[c];
        outF[(size_t)row * D_ + c] = v;
        if (outB) outB[(size_t)row * D_ + c] = f2b(v);
    }
}

// ---------------------------------------------------------------------------
// Launcher
// ---------------------------------------------------------------------------
extern "C" void kernel_launch(void* const* d_in, const int* in_sizes, int n_in,
                              void* d_out, int out_size, void* d_ws, size_t ws_size,
                              hipStream_t stream) {
    const float* x     = (const float*)d_in[0];
    const float* W_qkv = (const float*)d_in[1];
    const float* W_o   = (const float*)d_in[2];
    const float* b_o   = (const float*)d_in[3];
    const float* g1    = (const float*)d_in[4];
    const float* be1   = (const float*)d_in[5];
    const float* W1    = (const float*)d_in[6];
    const float* b1    = (const float*)d_in[7];
    const float* W2    = (const float*)d_in[8];
    const float* b2    = (const float*)d_in[9];
    const float* g2    = (const float*)d_in[10];
    const float* be2   = (const float*)d_in[11];
    float* out = (float*)d_out;

    // Workspace carve-up (256B aligned)
    char* ws = (char*)d_ws;
    size_t off = 0;
    auto alloc = [&](size_t bytes) -> char* {
        char* p = ws + off;
        off = (off + bytes + 255) & ~(size_t)255;
        return p;
    };
    unsigned short* xb    = (unsigned short*)alloc((size_t)TOK_ * D_ * 2);   // also reused as x1 bf16
    unsigned short* wqkvt = (unsigned short*)alloc((size_t)D3_ * D_ * 2);
    unsigned short* wot   = (unsigned short*)alloc((size_t)D_ * D_ * 2);
    unsigned short* w1t   = (unsigned short*)alloc((size_t)MLP_ * D_ * 2);
    unsigned short* w2t   = (unsigned short*)alloc((size_t)D_ * MLP_ * 2);
    unsigned short* qkv   = (unsigned short*)alloc((size_t)TOK_ * D3_ * 2);
    unsigned short* attn  = (unsigned short*)alloc((size_t)TOK_ * D_ * 2);
    float*          ybuf  = (float*)alloc((size_t)TOK_ * D_ * 4);            // reused as z
    float*          x1    = (float*)alloc((size_t)TOK_ * D_ * 4);
    unsigned short* hbuf  = (unsigned short*)alloc((size_t)TOK_ * MLP_ * 2);
    (void)ws_size; (void)in_sizes; (void)n_in; (void)out_size;

    // 1) casts / transposes
    cast_f32_bf16<<<(TOK_ * D_) / (256 * 4), 256, 0, stream>>>(x, xb, TOK_ * D_);
    transpose_cast<<<dim3(D3_ / 32, D_ / 32), dim3(32, 8), 0, stream>>>(W_qkv, wqkvt, D_, D3_);
    transpose_cast<<<dim3(D_ / 32,  D_ / 32), dim3(32, 8), 0, stream>>>(W_o,   wot,   D_, D_);
    transpose_cast<<<dim3(MLP_ / 32, D_ / 32), dim3(32, 8), 0, stream>>>(W1,   w1t,   D_, MLP_);
    transpose_cast<<<dim3(D_ / 32, MLP_ / 32), dim3(32, 8), 0, stream>>>(W2,   w2t,   MLP_, D_);

    // 2) qkv = x @ W_qkv  (Q columns pre-scaled by 0.125) -> bf16
    gemm_bf16<3><<<dim3(D3_ / 128, TOK_ / 128), 256, 0, stream>>>(
        xb, wqkvt, TOK_, D3_, D_, nullptr, nullptr, nullptr, qkv);

    // 3) attention
    attn_kernel<<<dim3(N_ / 128, B_ * H_), 128, 0, stream>>>(qkv, attn);

    // 4) y = attn @ W_o + b_o + x   -> fp32
    gemm_bf16<1><<<dim3(D_ / 128, TOK_ / 128), 256, 0, stream>>>(
        attn, wot, TOK_, D_, D_, b_o, x, ybuf, nullptr);

    // 5) x1 = LN(y)                 -> fp32 + bf16 (into xb)
    ln_kernel<<<TOK_, 256, 0, stream>>>(ybuf, g1, be1, x1, xb);

    // 6) h = gelu(x1 @ W1 + b1)     -> bf16
    gemm_bf16<2><<<dim3(MLP_ / 128, TOK_ / 128), 256, 0, stream>>>(
        xb, w1t, TOK_, MLP_, D_, b1, nullptr, nullptr, hbuf);

    // 7) z = h @ W2 + b2 + x1       -> fp32 (reuse ybuf)
    gemm_bf16<1><<<dim3(D_ / 128, TOK_ / 128), 256, 0, stream>>>(
        hbuf, w2t, TOK_, D_, MLP_, b2, x1, ybuf, nullptr);

    // 8) out = LN(z)
    ln_kernel<<<TOK_, 256, 0, stream>>>(ybuf, g2, be2, out, nullptr);
}
